// MultiHeaderAttention_38147899523762
// MI455X (gfx1250) — matrix operand order repaired and checked by reading, not compiled, not run on hardware
//
#include <hip/hip_runtime.h>
#include <hip/hip_bf16.h>

// ---------------------------------------------------------------------------
// Multi-head attention block for MI455X (gfx1250, wave32, WMMA).
// Pipeline: LN(f32->f16) -> QKV proj (f16 WMMA, async-LDS B tiles) ->
// causal flash-attention (f16 WMMA + online softmax) -> output proj (WMMA).
// Workspace usage: 64 MB of d_ws (staging f16 tensors + f16 weights).
// ---------------------------------------------------------------------------

#define DEVINL __device__ __forceinline__

typedef _Float16 v16h __attribute__((ext_vector_type(16)));
typedef _Float16 v2h  __attribute__((ext_vector_type(2)));
typedef float    v8f  __attribute__((ext_vector_type(8)));
typedef float    v4f  __attribute__((ext_vector_type(4)));

namespace mha {
constexpr int Bz   = 2;
constexpr int Sq   = 2048;
constexpr int IS   = 1024;
constexpr int H    = 16;
constexpr int HD   = 64;
constexpr int D    = 1024;   // H*HD
constexpr int ROWS = Bz * Sq; // 4096
}

DEVINL v8f zero8() {
  v8f z;
#pragma unroll
  for (int i = 0; i < 8; ++i) z[i] = 0.0f;
  return z;
}

// A-fragment (16x32 f16, MxK) per CDNA5 ISA layout:
// lane L: m = L&15, k-half = L>>4; VGPR j<4 holds K pair (2j + 8*kh),
// VGPR j>=4 holds K pair (2j+8 + 8*kh). Per lane this is two contiguous
// 8-half runs -> compiler emits two b128 loads.
DEVINL v16h load_a_frag(const _Float16* p, int ld, int lane) {
  const int m  = lane & 15;
  const int kh = (lane >> 4) << 3; // 0 or 8
  const _Float16* r = p + m * ld + kh;
  v16h a;
#pragma unroll
  for (int j = 0; j < 8; ++j) {
    const int kb = (j < 4) ? (2 * j) : (2 * j + 8);
    v2h t = *(const v2h*)(r + kb);
    a[2 * j]     = t.x;
    a[2 * j + 1] = t.y;
  }
  return a;
}

// B-fragment (32x16 f16, KxN): lane L holds column n = L&15; element s holds
// K = 8*(L>>4) + s for s<8, K = 16 + 8*(L>>4) + (s-8) for s>=8 (matches A).
DEVINL v16h load_b_frag(const _Float16* p, int ld, int lane) {
  const int n  = lane & 15;
  const int kh = (lane >> 4) << 3;
  v16h b;
#pragma unroll
  for (int s = 0; s < 16; ++s) {
    const int kk = (s < 8) ? (kh + s) : (16 + kh + s - 8);
    b[s] = p[(size_t)kk * ld + n];
  }
  return b;
}

DEVINL v8f wmma_f16(v16h a, v16h b, v8f c) {
  return __builtin_amdgcn_wmma_f32_16x16x32_f16(
      /*neg_a=*/false, a, /*neg_b=*/false, b,
      /*c_mod=*/(short)0, c, /*reuse_a=*/false, /*reuse_b=*/false);
}

// ---------------------------------------------------------------------------
// LayerNorm: one 256-thread block per row of 1024, fp32 in -> f16 out.
// ---------------------------------------------------------------------------
__global__ __launch_bounds__(256) void ln_f16_kernel(
    const float* __restrict__ x, const float* __restrict__ g,
    const float* __restrict__ bta, _Float16* __restrict__ y) {
  using namespace mha;
  const int row  = blockIdx.x;
  const int tid  = threadIdx.x;
  const int wave = tid >> 5;
  const int lane = tid & 31;
  const float* xr = x + (size_t)row * IS;
  v4f val = *(const v4f*)(xr + tid * 4);
  float s  = val[0] + val[1] + val[2] + val[3];
  float s2 = val[0]*val[0] + val[1]*val[1] + val[2]*val[2] + val[3]*val[3];
#pragma unroll
  for (int off = 1; off < 32; off <<= 1) {
    s  += __shfl_xor(s,  off, 32);
    s2 += __shfl_xor(s2, off, 32);
  }
  __shared__ float ws1[8], ws2[8];
  if (lane == 0) { ws1[wave] = s; ws2[wave] = s2; }
  __syncthreads();
  float ts = 0.0f, ts2 = 0.0f;
#pragma unroll
  for (int i = 0; i < 8; ++i) { ts += ws1[i]; ts2 += ws2[i]; }
  const float mu   = ts * (1.0f / IS);
  const float var  = ts2 * (1.0f / IS) - mu * mu;
  const float rstd = rsqrtf(var + 1e-5f);
  _Float16* yr = y + (size_t)row * IS;
#pragma unroll
  for (int i = 0; i < 4; ++i) {
    const int c = tid * 4 + i;
    yr[c] = (_Float16)((val[i] - mu) * rstd * g[c] + bta[c]);
  }
}

// ---------------------------------------------------------------------------
// fp32 -> f16 weight conversion.
// ---------------------------------------------------------------------------
__global__ __launch_bounds__(256) void f32_to_f16_kernel(
    const float* __restrict__ x, _Float16* __restrict__ y, int n) {
  const int i = blockIdx.x * 256 + threadIdx.x;
  if (i < n) y[i] = (_Float16)x[i];
}

// ---------------------------------------------------------------------------
// GEMM: C[4096,1024] = A_f16 @ W_f16 + bias, * row mask.
// 256 threads (8 waves), block tile 128x128, k-step 32.
// The 32x128 B tile is staged in LDS once per block via CDNA5 async LDS
// loads (global_load_async_to_lds_b128, ASYNCcnt), killing the 8x redundant
// per-wave global B traffic. Each wave computes 16 rows x 128 cols.
// mode 0: f16 out [B,H,S,HD] (Q,V); mode 1: f16 out [B,H,HD,S] (K^T);
// mode 3: f32 out row-major [4096,1024].
// ---------------------------------------------------------------------------
__global__ __launch_bounds__(256) void gemm128_kernel(
    const _Float16* __restrict__ A, const _Float16* __restrict__ W16,
    const float* __restrict__ bias, const float* __restrict__ rmask,
    _Float16* __restrict__ outh, float* __restrict__ outf, int mode) {
  using namespace mha;
  constexpr int LDB = 136;               // 128 + 8 halfs pad
  __shared__ _Float16 Bt[32 * LDB];
  const int tid  = threadIdx.x;
  const int lane = tid & 31;
  const int wave = tid >> 5;
  const int n0 = blockIdx.x * 128;
  const int m0 = blockIdx.y * 128 + wave * 16;

  v8f acc[8] = {zero8(), zero8(), zero8(), zero8(),
                zero8(), zero8(), zero8(), zero8()};

  // Cooperative async-load mapping: 256 threads x two 16B chunks cover the
  // 32x128 f16 tile. Thread t loads rows (t>>4) and (t>>4)+16.
  const int r0 = tid >> 4;               // 0..15
  const int c0 = (tid & 15) * 8;         // half offset within row
  const unsigned ldsA = (unsigned)(uintptr_t)&Bt[r0 * LDB + c0];
  const unsigned ldsB = (unsigned)(uintptr_t)&Bt[(r0 + 16) * LDB + c0];

  for (int k = 0; k < IS; k += 32) {
    __syncthreads();                     // previous tile fully consumed
    const _Float16* src = W16 + (size_t)(k + r0) * D + n0 + c0;
    const unsigned long long g0 = (unsigned long long)(uintptr_t)src;
    const unsigned long long g1 = (unsigned long long)(uintptr_t)(src + (size_t)16 * D);
    asm volatile("global_load_async_to_lds_b128 %0, %1, off"
                 :: "v"(ldsA), "v"(g0) : "memory");
    asm volatile("global_load_async_to_lds_b128 %0, %1, off"
                 :: "v"(ldsB), "v"(g1) : "memory");
    const v16h a = load_a_frag(A + (size_t)m0 * IS + k, IS, lane);
    asm volatile("s_wait_asynccnt 0x0" ::: "memory");
    __syncthreads();                     // tile visible to all waves
#pragma unroll
    for (int nb = 0; nb < 8; ++nb) {
      const v16h bf = load_b_frag(&Bt[nb * 16], LDB, lane);
      acc[nb] = wmma_f16(a, bf, acc[nb]);
    }
  }

  const int half = lane >> 4;
  const int col  = lane & 15;
  float rm[8];
#pragma unroll
  for (int v = 0; v < 8; ++v) rm[v] = rmask[m0 + v + 8 * half];
#pragma unroll
  for (int nb = 0; nb < 8; ++nb) {
    const int c  = n0 + nb * 16 + col;
    const float bv = bias[c];
    const int hh = c >> 6;      // head
    const int dd = c & 63;      // within-head dim
#pragma unroll
    for (int v = 0; v < 8; ++v) {
      const int r  = m0 + v + 8 * half;
      const float val = (acc[nb][v] + bv) * rm[v];
      if (mode == 3) {
        outf[(size_t)r * D + c] = val;
      } else {
        const int bb = r >> 11;   // batch (S=2048)
        const int ss = r & 2047;
        size_t idx;
        if (mode == 0) idx = (((size_t)(bb * H + hh) * Sq + ss) * HD + dd);
        else           idx = (((size_t)(bb * H + hh) * HD + dd) * Sq + ss);
        outh[idx] = (_Float16)val;
      }
    }
  }
}

// ---------------------------------------------------------------------------
// Causal flash attention. Grid: (S/64, B*H), 128 threads (4 waves).
// Each wave owns 16 query rows; streams keys in blocks of 32.
// ---------------------------------------------------------------------------
__global__ __launch_bounds__(128) void attn_kernel(
    const _Float16* __restrict__ Q,   // [B,H,S,HD]
    const _Float16* __restrict__ Kt,  // [B,H,HD,S]
    const _Float16* __restrict__ V,   // [B,H,S,HD]
    _Float16* __restrict__ Aout) {    // [B,S,D]
  using namespace mha;
  __shared__ _Float16 Plds[4][16 * 34];
  const int lane = threadIdx.x & 31;
  const int wave = threadIdx.x >> 5;
  const int bh = blockIdx.y;
  const int bb = bh >> 4;
  const int hh = bh & 15;
  const int q0 = blockIdx.x * 64;
  const int mrow = q0 + wave * 16;

  const _Float16* Qh  = Q  + ((size_t)bh * Sq + mrow) * HD;
  const _Float16* Kth = Kt + (size_t)bh * HD * Sq;
  const _Float16* Vh  = V  + (size_t)bh * Sq * HD;

  const v16h aq0 = load_a_frag(Qh,      HD, lane);  // K dims 0..31
  const v16h aq1 = load_a_frag(Qh + 32, HD, lane);  // K dims 32..63

  v8f o[4] = {zero8(), zero8(), zero8(), zero8()};
  float mx[8], ls[8];
#pragma unroll
  for (int v = 0; v < 8; ++v) { mx[v] = -1e30f; ls[v] = 0.0f; }

  const int half = lane >> 4;
  const int col  = lane & 15;
  const float scale = 0.125f;  // 1/sqrt(64)
  _Float16* Pw = &Plds[wave][0];

  const int jend = q0 + 64;    // causal: keys beyond query tile are all masked
  for (int j = 0; j < jend; j += 32) {
    // ---- S = Q K^T for 16 q-rows x 32 keys (two 16x16 fragments) ----
    v8f s0 = zero8(), s1 = zero8();
    s0 = wmma_f16(aq0, load_b_frag(Kth + j,                Sq, lane), s0);
    s0 = wmma_f16(aq1, load_b_frag(Kth + 32 * Sq + j,      Sq, lane), s0);
    s1 = wmma_f16(aq0, load_b_frag(Kth + j + 16,           Sq, lane), s1);
    s1 = wmma_f16(aq1, load_b_frag(Kth + 32 * Sq + j + 16, Sq, lane), s1);

    // ---- causal mask + online softmax (row stats live across 16 lanes) ----
#pragma unroll
    for (int v = 0; v < 8; ++v) {
      const int r  = mrow + v + 8 * half;   // query position
      const int c0 = j + col;
      const int c1 = j + 16 + col;
      float x0 = s0[v] * scale - (c0 > r ? 1e9f : 0.0f);
      float x1 = s1[v] * scale - (c1 > r ? 1e9f : 0.0f);
      float m = fmaxf(mx[v], fmaxf(x0, x1));
#pragma unroll
      for (int off = 1; off < 16; off <<= 1) m = fmaxf(m, __shfl_xor(m, off, 32));
      const float corr = __expf(mx[v] - m);
      const float p0 = __expf(x0 - m);
      const float p1 = __expf(x1 - m);
      float t = p0 + p1;
#pragma unroll
      for (int off = 1; off < 16; off <<= 1) t += __shfl_xor(t, off, 32);
      ls[v] = ls[v] * corr + t;
      mx[v] = m;
      o[0][v] *= corr; o[1][v] *= corr; o[2][v] *= corr; o[3][v] *= corr;
      const int lr = v + 8 * half;
      Pw[lr * 34 + col]      = (_Float16)p0;
      Pw[lr * 34 + 16 + col] = (_Float16)p1;
    }

    // LDS RAW: DS is in-order per wave; fence the compiler + drain dscnt.
    asm volatile("s_wait_dscnt 0x0" ::: "memory");
    const v16h ap = load_a_frag(Pw, 34, lane);   // P in A-fragment layout
    asm volatile("" ::: "memory");               // keep loads before next stores

    // ---- O += P @ V ----
#pragma unroll
    for (int nb = 0; nb < 4; ++nb) {
      const v16h bv = load_b_frag(Vh + (size_t)j * HD + nb * 16, HD, lane);
      o[nb] = wmma_f16(ap, bv, o[nb]);
    }
  }

  // ---- finalize: divide by row sum, store f16 [B,S,D] ----
  float inv[8];
#pragma unroll
  for (int v = 0; v < 8; ++v) inv[v] = 1.0f / ls[v];
#pragma unroll
  for (int nb = 0; nb < 4; ++nb) {
    const int c = hh * HD + nb * 16 + col;
#pragma unroll
    for (int v = 0; v < 8; ++v) {
      const int ss = mrow + v + 8 * half;
      Aout[((size_t)(bb * Sq + ss)) * D + c] = (_Float16)(o[nb][v] * inv[v]);
    }
  }
}

// ---------------------------------------------------------------------------
extern "C" void kernel_launch(void* const* d_in, const int* in_sizes, int n_in,
                              void* d_out, int out_size, void* d_ws, size_t ws_size,
                              hipStream_t stream) {
  using namespace mha;
  (void)in_sizes; (void)n_in; (void)out_size; (void)ws_size;
  const float* q       = (const float*)d_in[0];
  const float* k       = (const float*)d_in[1];
  const float* v       = (const float*)d_in[2];
  const float* q_mask  = (const float*)d_in[3];   // [B*S]
  const float* kv_mask = (const float*)d_in[4];   // [B*S]
  // d_in[5] = a_mask: causal penalty, reproduced analytically in attn_kernel.
  const float* ln_g = (const float*)d_in[6];
  const float* ln_b = (const float*)d_in[7];
  const float* Wq = (const float*)d_in[8];
  const float* bq = (const float*)d_in[9];
  const float* Wk = (const float*)d_in[10];
  const float* bk = (const float*)d_in[11];
  const float* Wv = (const float*)d_in[12];
  const float* bv = (const float*)d_in[13];
  const float* Wo = (const float*)d_in[14];
  const float* bo = (const float*)d_in[15];
  float* out = (float*)d_out;

  char* ws = (char*)d_ws;
  const size_t MB = 1024u * 1024u;
  _Float16* qn   = (_Float16*)(ws + 0 * MB);   // [4096,1024] f16
  _Float16* kn   = (_Float16*)(ws + 8 * MB);
  _Float16* vn   = (_Float16*)(ws + 16 * MB);
  _Float16* wq16 = (_Float16*)(ws + 24 * MB);  // [1024,1024] f16
  _Float16* wk16 = (_Float16*)(ws + 26 * MB);
  _Float16* wv16 = (_Float16*)(ws + 28 * MB);
  _Float16* wo16 = (_Float16*)(ws + 30 * MB);
  _Float16* Qb   = (_Float16*)(ws + 32 * MB);  // [B,H,S,HD]
  _Float16* Ktb  = (_Float16*)(ws + 40 * MB);  // [B,H,HD,S]
  _Float16* Vb   = (_Float16*)(ws + 48 * MB);  // [B,H,S,HD]
  _Float16* Ao   = (_Float16*)(ws + 56 * MB);  // [B,S,D] f16

  // 1) LayerNorm to f16
  ln_f16_kernel<<<ROWS, 256, 0, stream>>>(q, ln_g, ln_b, qn);
  ln_f16_kernel<<<ROWS, 256, 0, stream>>>(k, ln_g, ln_b, kn);
  ln_f16_kernel<<<ROWS, 256, 0, stream>>>(v, ln_g, ln_b, vn);

  // 2) Weights to f16
  const int WN = IS * D;
  f32_to_f16_kernel<<<WN / 256, 256, 0, stream>>>(Wq, wq16, WN);
  f32_to_f16_kernel<<<WN / 256, 256, 0, stream>>>(Wk, wk16, WN);
  f32_to_f16_kernel<<<WN / 256, 256, 0, stream>>>(Wv, wv16, WN);
  f32_to_f16_kernel<<<WN / 256, 256, 0, stream>>>(Wo, wo16, WN);

  // 3) Q/K/V projections (WMMA + async-LDS B tiles).
  //    K written transposed so QK^T B-fragments are contiguous loads.
  dim3 pg(D / 128, ROWS / 128);
  gemm128_kernel<<<pg, 256, 0, stream>>>(qn, wq16, bq, q_mask,  Qb,  nullptr, 0);
  gemm128_kernel<<<pg, 256, 0, stream>>>(kn, wk16, bk, kv_mask, Ktb, nullptr, 1);
  gemm128_kernel<<<pg, 256, 0, stream>>>(vn, wv16, bv, kv_mask, Vb,  nullptr, 0);

  // 4) Causal flash attention (WMMA + online softmax)
  dim3 ag(Sq / 64, Bz * H);
  attn_kernel<<<ag, 128, 0, stream>>>(Qb, Ktb, Vb, Ao);

  // 5) Output projection (WMMA, fp32 out)
  gemm128_kernel<<<pg, 256, 0, stream>>>(Ao, wo16, bo, kv_mask, nullptr, out, 3);
}
